// GINConvBlock_63780264345859
// MI455X (gfx1250) — matrix operand order repaired and checked by reading, not compiled, not run on hardware
//
#include <hip/hip_runtime.h>

#define D 128

typedef __attribute__((ext_vector_type(16))) _Float16 v16h;
typedef __attribute__((ext_vector_type(8)))  _Float16 v8h;
typedef __attribute__((ext_vector_type(8)))  float    v8f;
typedef __attribute__((ext_vector_type(4)))  float    v4f;

// h0 = x  (seeds agg with x so that h = x + agg falls out of the scatter), zero BN stats
__global__ void init_h0_kernel(const float* __restrict__ x, float* __restrict__ h0,
                               float* __restrict__ sums, int total) {
    int idx = blockIdx.x * blockDim.x + threadIdx.x;
    if (idx < 2 * D) sums[idx] = 0.0f;
    if (idx < total) h0[idx] = x[idx];
}

// W[k][n] (row-major f32) -> Wt[n][k] (column-major f16): B-fragment reads become contiguous
__global__ void cvt_weight_kernel(const float* __restrict__ W, _Float16* __restrict__ Wt) {
    int idx = blockIdx.x * blockDim.x + threadIdx.x;   // 0 .. D*D-1
    int k = idx >> 7;
    int n = idx & (D - 1);
    Wt[n * D + k] = (_Float16)W[k * D + n];
}

// GIN aggregation: h0[dst] += x[src]. 32 lanes x 4 feats per edge; edge ids are
// wave-uniform so the index loads scalarize; x row reads are b128.
__global__ void scatter_kernel(const float* __restrict__ x, const int* __restrict__ ei,
                               float* __restrict__ h0, int E_) {
    int e = blockIdx.x * 8 + (threadIdx.x >> 5);       // uniform within a wave32
    if (e >= E_) return;
    e = __builtin_amdgcn_readfirstlane(e);
    int src = ei[e];                                   // edge_index row 0 (scalar load)
    int dst = ei[E_ + e];                              // edge_index row 1 (scalar load)
    int f0 = (threadIdx.x & 31) * 4;
    v4f v = *(const v4f*)(x + (size_t)src * D + f0);
    float* p = h0 + (size_t)dst * D + f0;
    atomicAdd(p + 0, v[0]);
    atomicAdd(p + 1, v[1]);
    atomicAdd(p + 2, v[2]);
    atomicAdd(p + 3, v[3]);
}

// ---- fragment load helpers ---------------------------------------------------

// A fragment from f32 row-major source (converted to f16)
__device__ __forceinline__ v16h load_a_f32(const float* ap, int k) {
    v4f a0 = *(const v4f*)(ap + k);
    v4f a1 = *(const v4f*)(ap + k + 4);
    v4f a2 = *(const v4f*)(ap + k + 8);
    v4f a3 = *(const v4f*)(ap + k + 12);
    v16h a;
#pragma unroll
    for (int j = 0; j < 4; ++j) {
        a[j]      = (_Float16)a0[j];
        a[4 + j]  = (_Float16)a1[j];
        a[8 + j]  = (_Float16)a2[j];
        a[12 + j] = (_Float16)a3[j];
    }
    return a;
}

// A fragment from f16 row-major source (two b128 loads, no conversion)
__device__ __forceinline__ v16h load_a_f16(const _Float16* ap, int k) {
    v8h lo = *(const v8h*)(ap + k);
    v8h hh = *(const v8h*)(ap + k + 8);
    v16h a;
#pragma unroll
    for (int j = 0; j < 8; ++j) {
        a[j]     = lo[j];
        a[8 + j] = hh[j];
    }
    return a;
}

// GEMM1: h1[N x 128] = relu(h0 @ W1 + b1), f32 in, f16 out.
// One wave = 16 rows x 64 cols (4 column tiles, 4 accumulators), 4 waves/block.
// sched_barrier(0) pins all B-tile (+ next-A) loads before the WMMA chain so the
// B fragments stay in distinct registers and waits retire progressively.
// __launch_bounds__(128, 1): allow the full live set in VGPRs (no scratch spill).
__global__ void __launch_bounds__(128, 1)
wmma_gemm1_kernel(const float* __restrict__ A, const _Float16* __restrict__ Bt,
                  const float* __restrict__ bias, _Float16* __restrict__ Cout,
                  int Nrows) {
    const int wave  = threadIdx.x >> 5;
    const int lane  = threadIdx.x & 31;
    const int m0    = (blockIdx.x * 4 + wave) * 16;
    const int nbase = blockIdx.y * 64;
    const int l16   = lane & 15;
    const int hi    = lane >> 4;             // 0: lanes 0-15, 1: lanes 16-31
    int rowA = m0 + l16;
    if (rowA >= Nrows) rowA = Nrows - 1;     // clamp: loads in-bounds, store is guarded
    const int koffA = hi * 16;               // contiguous A layout (upper lanes: K+16..31)
    const int koffB = hi * 16;               // contiguous B layout, same as A (upper lanes: K+16..31)

    v8f acc[4];
#pragma unroll
    for (int t = 0; t < 4; ++t) {
        float bv = bias[nbase + t * 16 + l16];
        acc[t] = (v8f){bv, bv, bv, bv, bv, bv, bv, bv};
    }

    const float*    ap = A  + (size_t)rowA * D + koffA;
    const _Float16* bp = Bt + (size_t)(nbase + l16) * D + koffB;

    v16h a = load_a_f32(ap, 0);
#pragma unroll
    for (int k = 0; k < D; k += 32) {
        v16h bf[4];
#pragma unroll
        for (int t = 0; t < 4; ++t)
            bf[t] = *(const v16h*)(bp + (size_t)t * 16 * D + k);   // grouped load clause
        v16h an;
        if (k + 32 < D) an = load_a_f32(ap, k + 32);               // pipeline next A
        __builtin_amdgcn_sched_barrier(0);   // all loads above issue before WMMAs below
#pragma unroll
        for (int t = 0; t < 4; ++t)
            acc[t] = __builtin_amdgcn_wmma_f32_16x16x32_f16(false, a, false, bf[t],
                                                            (short)0, acc[t], false, false);
        a = an;
    }

#pragma unroll
    for (int t = 0; t < 4; ++t) {
#pragma unroll
        for (int r = 0; r < 8; ++r) {
            int row = m0 + r + hi * 8;       // C layout: lanes 16-31 hold rows M=8..15
            if (row < Nrows)
                Cout[(size_t)row * D + nbase + t * 16 + l16] =
                    (_Float16)fmaxf(acc[t][r], 0.0f);
        }
    }
}

// GEMM2: h2[N x 128] = h1 @ W2 + b2, f16 in (no conversions in loop), f32 out.
__global__ void __launch_bounds__(128, 1)
wmma_gemm2_kernel(const _Float16* __restrict__ A, const _Float16* __restrict__ Bt,
                  const float* __restrict__ bias, float* __restrict__ Cout,
                  int Nrows) {
    const int wave  = threadIdx.x >> 5;
    const int lane  = threadIdx.x & 31;
    const int m0    = (blockIdx.x * 4 + wave) * 16;
    const int nbase = blockIdx.y * 64;
    const int l16   = lane & 15;
    const int hi    = lane >> 4;
    int rowA = m0 + l16;
    if (rowA >= Nrows) rowA = Nrows - 1;
    const int koffA = hi * 16;
    const int koffB = hi * 16;

    v8f acc[4];
#pragma unroll
    for (int t = 0; t < 4; ++t) {
        float bv = bias[nbase + t * 16 + l16];
        acc[t] = (v8f){bv, bv, bv, bv, bv, bv, bv, bv};
    }

    const _Float16* ap = A  + (size_t)rowA * D + koffA;
    const _Float16* bp = Bt + (size_t)(nbase + l16) * D + koffB;

    v16h a = load_a_f16(ap, 0);
#pragma unroll
    for (int k = 0; k < D; k += 32) {
        v16h bf[4];
#pragma unroll
        for (int t = 0; t < 4; ++t)
            bf[t] = *(const v16h*)(bp + (size_t)t * 16 * D + k);
        v16h an;
        if (k + 32 < D) an = load_a_f16(ap, k + 32);
        __builtin_amdgcn_sched_barrier(0);
#pragma unroll
        for (int t = 0; t < 4; ++t)
            acc[t] = __builtin_amdgcn_wmma_f32_16x16x32_f16(false, a, false, bf[t],
                                                            (short)0, acc[t], false, false);
        a = an;
    }

#pragma unroll
    for (int t = 0; t < 4; ++t) {
#pragma unroll
        for (int r = 0; r < 8; ++r) {
            int row = m0 + r + hi * 8;
            if (row < Nrows)
                Cout[(size_t)row * D + nbase + t * 16 + l16] = acc[t][r];
        }
    }
}

// Per-column sum / sum-of-squares partials; one atomic per block per column.
__global__ void stats_kernel(const float* __restrict__ h2, float* __restrict__ sums, int Nrows) {
    int col = threadIdx.x;               // blockDim.x == D
    float s = 0.0f, s2 = 0.0f;
    for (int row = blockIdx.x; row < Nrows; row += gridDim.x) {
        float v = h2[(size_t)row * D + col];
        s  += v;
        s2 += v * v;
    }
    atomicAdd(&sums[col], s);
    atomicAdd(&sums[D + col], s2);
}

// BatchNorm (batch stats) + ReLU + residual, fused
__global__ void finalize_kernel(const float* __restrict__ h2, const float* __restrict__ sums,
                                const float* __restrict__ gamma, const float* __restrict__ beta,
                                const float* __restrict__ residual, float* __restrict__ out,
                                int Nrows) {
    int idx = blockIdx.x * blockDim.x + threadIdx.x;
    int total = Nrows * D;
    if (idx >= total) return;
    int col = idx & (D - 1);
    float invN = 1.0f / (float)Nrows;
    float mean = sums[col] * invN;
    float var  = sums[D + col] * invN - mean * mean;
    float v = (h2[idx] - mean) * rsqrtf(var + 1e-5f) * gamma[col] + beta[col];
    v = fmaxf(v, 0.0f);
    out[idx] = v + residual[idx];
}

extern "C" void kernel_launch(void* const* d_in, const int* in_sizes, int n_in,
                              void* d_out, int out_size, void* d_ws, size_t ws_size,
                              hipStream_t stream) {
    const float* x        = (const float*)d_in[0];
    const int*   ei       = (const int*)  d_in[1];
    const float* residual = (const float*)d_in[2];
    const float* W1       = (const float*)d_in[3];
    const float* b1       = (const float*)d_in[4];
    const float* W2       = (const float*)d_in[5];
    const float* b2       = (const float*)d_in[6];
    const float* gamma    = (const float*)d_in[7];
    const float* beta     = (const float*)d_in[8];

    const int N_ = in_sizes[0] / D;      // 50000
    const int E_ = in_sizes[1] / 2;      // 800000
    const size_t nd = (size_t)N_ * D;

    char* ws = (char*)d_ws;
    float*    h0   = (float*)ws;                              // nd f32 (reused as h2)
    _Float16* h1   = (_Float16*)(ws + nd * sizeof(float));    // nd f16
    _Float16* Wt1  = (_Float16*)(ws + nd * sizeof(float) + nd * sizeof(_Float16));
    _Float16* Wt2  = Wt1 + D * D;
    float*    sums = (float*)(Wt2 + D * D);                   // 2*D f32
    float*    h2   = h0;                                      // gemm2 reads h1, writes h0

    const int total = (int)nd;
    {
        int blocks = (total + 255) / 256;
        hipLaunchKernelGGL(init_h0_kernel, dim3(blocks), dim3(256), 0, stream, x, h0, sums, total);
    }
    hipLaunchKernelGGL(cvt_weight_kernel, dim3((D * D) / 256), dim3(256), 0, stream, W1, Wt1);
    hipLaunchKernelGGL(cvt_weight_kernel, dim3((D * D) / 256), dim3(256), 0, stream, W2, Wt2);
    {
        int blocks = (E_ + 7) / 8;       // 8 edges per 256-thread block
        hipLaunchKernelGGL(scatter_kernel, dim3(blocks), dim3(256), 0, stream, x, ei, h0, E_);
    }
    dim3 ggrid((N_ + 63) / 64, 2);       // 4 waves x 16 rows, 2 x 64-col panels
    hipLaunchKernelGGL(wmma_gemm1_kernel, ggrid, dim3(128), 0, stream, h0, Wt1, b1, h1, N_);
    hipLaunchKernelGGL(wmma_gemm2_kernel, ggrid, dim3(128), 0, stream, h1, Wt2, b2, h2, N_);
    hipLaunchKernelGGL(stats_kernel, dim3(512), dim3(D), 0, stream, h2, sums, N_);
    {
        int blocks = (total + 255) / 256;
        hipLaunchKernelGGL(finalize_kernel, dim3(blocks), dim3(256), 0, stream,
                           h2, sums, gamma, beta, residual, (float*)d_out, N_);
    }
}
